// GraphAttentionLayer_76888504533416
// MI455X (gfx1250) — matrix operand order repaired and checked by reading, not compiled, not run on hardware
//
#include <hip/hip_runtime.h>
#include <hip/hip_bf16.h>

// GAT layer for MI455X (gfx1250). Strategy:
//  - Never materialize E=emb[adj] (1GB). Decompose scores & aggregation:
//      s[b,i,j]   = lrelu(ha1[b,i] + ha2[b,j] + ea2[adj] + a_b)
//      out[b,i,:] = attn @ h  +  wsum @ emb   (wsum: per-edge-type attn mass)
//  - Both GEMMs (typed projection, attn@h) run on v_wmma_f32_16x16x32_f16.
//  - Total HBM traffic ~25MB -> bandwidth bound at 23.3 TB/s; f16 matrix path
//    + f32 softmax is the right precision/bandwidth tradeoff.

typedef __attribute__((ext_vector_type(16))) _Float16 v16h;
typedef __attribute__((ext_vector_type(8)))  float    v8f;

#define WMMA_F16(a, b, c) \
  __builtin_amdgcn_wmma_f32_16x16x32_f16(false, (a), false, (b), (short)0, (c), false, false)

#define GAT_B     8
#define GAT_N     512
#define GAT_FIN   256
#define GAT_FOUT  128
#define GAT_T     3
#define GAT_NE1   11     // NE+1 embedding rows
#define GAT_ROWS  (GAT_B * GAT_N)   // 4096

// ---------------------------------------------------------------------------
// Kernel 0: transpose/convert W (3,256,128) f32 -> Wt (3,128,256) f16,
// and compute ea2[k] = emb[k] . a2 (11 scalars).
// ---------------------------------------------------------------------------
__global__ void gat_prep(const float* __restrict__ W, const float* __restrict__ emb,
                         const float* __restrict__ a2,
                         _Float16* __restrict__ Wt, float* __restrict__ ea2) {
  int idx = blockIdx.x * 256 + threadIdx.x;        // enumerates (t, n, k), k fastest
  int t   = idx / (GAT_FOUT * GAT_FIN);
  int rem = idx - t * (GAT_FOUT * GAT_FIN);
  int n   = rem >> 8;                              // /256
  int k   = rem & 255;
  Wt[idx] = (_Float16)W[(t * GAT_FIN + k) * GAT_FOUT + n];

  if (blockIdx.x == 0 && threadIdx.x < GAT_NE1) {
    float s = 0.f;
    const float* e = emb + threadIdx.x * GAT_FOUT;
    for (int f = 0; f < GAT_FOUT; ++f) s += e[f] * a2[f];
    ea2[threadIdx.x] = s;
  }
}

// ---------------------------------------------------------------------------
// Kernel 1: typed projection  h = sum_t mask_t * (X @ W_t + b_t)
// One wave -> one 16x16 output tile; 3 accumulators (one per node type).
// Writes h32 (4096,128) f32 and hT16 (8,128,512) f16 (transposed, j-contig).
// grid 256 x 256 threads (8 waves/block); K loop 256/32 = 8 WMMA steps x3.
// ---------------------------------------------------------------------------
__global__ void __launch_bounds__(256)
gat_proj(const float* __restrict__ X,            // (4096,256)
         const float* __restrict__ mask,         // (3,4096)  [(T,B,N,1) flattened]
         const _Float16* __restrict__ Wt,        // (3,128,256)
         const float* __restrict__ bias,         // (3,128)
         float* __restrict__ h32,                // (4096,128)
         _Float16* __restrict__ hT16) {          // (8,128,512)
  const int wid  = threadIdx.x >> 5;
  const int lane = threadIdx.x & 31;
  const int half = lane >> 4;
  const int l16  = lane & 15;

  int task  = blockIdx.x * 8 + wid;   // 2048 tasks = 256 mtiles * 8 ntiles
  int mtile = task >> 3;
  int ntile = task & 7;

  const float*    arow = X  + (size_t)(mtile * 16 + l16) * GAT_FIN;
  const _Float16* b0   = Wt + (size_t)(0 * GAT_FOUT + ntile * 16 + l16) * GAT_FIN;
  const _Float16* b1   = Wt + (size_t)(1 * GAT_FOUT + ntile * 16 + l16) * GAT_FIN;
  const _Float16* b2   = Wt + (size_t)(2 * GAT_FOUT + ntile * 16 + l16) * GAT_FIN;

  v8f acc0 = {}, acc1 = {}, acc2 = {};
  for (int ks = 0; ks < GAT_FIN / 32; ++ks) {
    int kb = ks * 32;
    v16h a, f0, f1, f2;
#pragma unroll
    for (int u = 0; u < 16; ++u) {                 // A layout: K-halves of 16
      a[u]     = (_Float16)arow[kb + half * 16 + u];
    }
#pragma unroll
    for (int u = 0; u < 16; ++u) {                 // B layout: K-halves of 16
      f0[u] = b0[kb + half * 16 + u];
      f1[u] = b1[kb + half * 16 + u];
      f2[u] = b2[kb + half * 16 + u];
    }
    acc0 = WMMA_F16(a, f0, acc0);
    acc1 = WMMA_F16(a, f1, acc1);
    acc2 = WMMA_F16(a, f2, acc2);
  }

  int ncol = ntile * 16 + l16;
#pragma unroll
  for (int v = 0; v < 8; ++v) {                    // C layout: M = v + 8*half
    int row = mtile * 16 + 8 * half + v;
    float m0 = mask[0 * GAT_ROWS + row];
    float m1 = mask[1 * GAT_ROWS + row];
    float m2 = mask[2 * GAT_ROWS + row];
    float hv = m0 * (acc0[v] + bias[0 * GAT_FOUT + ncol]) +
               m1 * (acc1[v] + bias[1 * GAT_FOUT + ncol]) +
               m2 * (acc2[v] + bias[2 * GAT_FOUT + ncol]);
    h32[(size_t)row * GAT_FOUT + ncol] = hv;
    int bn = row >> 9, nn = row & 511;
    hT16[((size_t)bn * GAT_FOUT + ncol) * GAT_N + nn] = (_Float16)hv;
  }
}

// ---------------------------------------------------------------------------
// Kernel 2: ha1[row] = h[row].a1 ; ha2[row] = h[row].a2  (wave per row,
// 4 elems/lane + shfl_xor reduction over the 32-lane wave).
// ---------------------------------------------------------------------------
__global__ void __launch_bounds__(256)
gat_rowdot(const float* __restrict__ h32, const float* __restrict__ a1,
           const float* __restrict__ a2, float* __restrict__ ha1,
           float* __restrict__ ha2) {
  int wid = threadIdx.x >> 5, lane = threadIdx.x & 31;
  int row = blockIdx.x * 8 + wid;
  const float* hr = h32 + (size_t)row * GAT_FOUT;
  float s1 = 0.f, s2 = 0.f;
#pragma unroll
  for (int u = 0; u < 4; ++u) {
    int c = lane * 4 + u;
    float hv = hr[c];
    s1 += hv * a1[c];
    s2 += hv * a2[c];
  }
#pragma unroll
  for (int off = 16; off >= 1; off >>= 1) {
    s1 += __shfl_xor(s1, off, 32);
    s2 += __shfl_xor(s2, off, 32);
  }
  if (lane == 0) { ha1[row] = s1; ha2[row] = s2; }
}

// ---------------------------------------------------------------------------
// Kernel 3: fused attention. Block = 128 threads (4 waves) per (b, 16-row tile).
//  Phase A: raw scores -> LDS (f32), per-group running max
//  Phase B: row max reduce; zero wsum
//  Phase C: e = exp(s - max) -> LDS; row-sum partials; wsum via LDS atomicAdd
//  Phase D: normalize (1/sum), scale wsum
//  Phase E: attn -> f16 LDS (A-fragment source)
//  GEMM   : out = attn @ hT  (16x128x512 via 2 n-tiles/wave x 16 K-steps),
//           + wsum @ emb epilogue, ELU, store f32.
// ---------------------------------------------------------------------------
__global__ void __launch_bounds__(128)
gat_attn(const int* __restrict__ adj,            // (8,512,512)
         const float* __restrict__ ha1,          // (4096)
         const float* __restrict__ ha2,          // (4096)
         const float* __restrict__ ea2,          // (11)
         const float* __restrict__ a_b,          // (1)
         const _Float16* __restrict__ hT16,      // (8,128,512)
         const float* __restrict__ emb,          // (11,128)
         float* __restrict__ out) {              // (8,512,128)
  __shared__ float    sf[16][GAT_N];             // 32 KB  scores / exp
  __shared__ _Float16 p16[16][GAT_N];            // 16 KB  attn f16 (A fragments)
  __shared__ float    red[16][8];
  __shared__ float    rowmax[16];
  __shared__ float    rowinv[16];
  __shared__ float    wsumS[16][12];             // 11 edge types (padded)

  const int b  = blockIdx.x >> 5;
  const int i0 = (blockIdx.x & 31) * 16;
  const int t  = threadIdx.x;
  const int g  = t >> 3;                         // row within tile (0..15)
  const int r  = t & 7;                          // 8 threads per row

  const float ab = a_b[0];
  const float h1 = ha1[b * GAT_N + i0 + g];
  const int* arow = adj + ((size_t)(b * GAT_N + i0 + g)) * GAT_N;

  // Phase A: raw scores
  float pmax = -3.0e38f;
  for (int jj = 0; jj < 64; ++jj) {
    int j  = r * 64 + jj;
    int av = arow[j];
    float s = h1 + ha2[b * GAT_N + j] + ea2[av] + ab;
    s = s > 0.f ? s : 0.2f * s;                  // leaky relu
    s = (av > 0) ? s : -9.0e15f;                 // mask
    sf[g][j] = s;
    pmax = fmaxf(pmax, s);
  }
  red[g][r] = pmax;
  __syncthreads();

  // Phase B: row max + zero wsum
  if (t < 16) {
    float m = red[t][0];
#pragma unroll
    for (int u = 1; u < 8; ++u) m = fmaxf(m, red[t][u]);
    rowmax[t] = m;
#pragma unroll
    for (int k = 0; k < 12; ++k) wsumS[t][k] = 0.f;
  }
  __syncthreads();

  // Phase C: exponentials, partial sums, per-edge-type attn mass
  float rm = rowmax[g];
  float psum = 0.f;
  for (int jj = 0; jj < 64; ++jj) {
    int j  = r * 64 + jj;
    int av = arow[j];
    float e = __expf(sf[g][j] - rm);
    sf[g][j] = e;
    psum += e;
    atomicAdd(&wsumS[g][av], e);                 // ds_add_f32
  }
  red[g][r] = psum;
  __syncthreads();

  // Phase D: normalize
  if (t < 16) {
    float s = 0.f;
#pragma unroll
    for (int u = 0; u < 8; ++u) s += red[t][u];
    float inv = 1.0f / s;
    rowinv[t] = inv;
#pragma unroll
    for (int k = 0; k < 11; ++k) wsumS[t][k] *= inv;
  }
  __syncthreads();

  // Phase E: normalized attn -> f16 LDS
  for (int c = 0; c < 64; ++c) {
    int idx = t + 128 * c;
    int i = idx >> 9, j = idx & 511;
    p16[i][j] = (_Float16)(sf[i][j] * rowinv[i]);
  }
  __syncthreads();

  // GEMM: out_tile(16x128) = attn(16x512) @ hT(512x128) via WMMA
  const int wv = t >> 5, lane = t & 31, half = lane >> 4, l16 = lane & 15;
#pragma unroll
  for (int nn = 0; nn < 2; ++nn) {
    int nt = wv + nn * 4;                        // 8 n-tiles over 4 waves
    const _Float16* bp = hT16 + ((size_t)(b * GAT_FOUT + nt * 16 + l16)) * GAT_N;
    v8f acc = {};
    for (int ks = 0; ks < GAT_N / 32; ++ks) {
      int kb = ks * 32;
      v16h a, bf;
#pragma unroll
      for (int u = 0; u < 16; ++u) {
        a[u]     = p16[l16][kb + half * 16 + u];
      }
#pragma unroll
      for (int u = 0; u < 16; ++u) bf[u] = bp[kb + half * 16 + u];
      acc = WMMA_F16(a, bf, acc);
    }
#pragma unroll
    for (int v = 0; v < 8; ++v) {
      int m    = v + 8 * half;
      int fcol = nt * 16 + l16;
      float o  = acc[v];
#pragma unroll
      for (int k = 0; k < 11; ++k) o += wsumS[m][k] * emb[k * GAT_FOUT + fcol];
      o = o > 0.f ? o : (__expf(o) - 1.0f);      // ELU
      out[((size_t)(b * GAT_N + i0 + m)) * GAT_FOUT + fcol] = o;
    }
  }
}

// ---------------------------------------------------------------------------
// Workspace layout (bytes): Wt f16 @0 (192KB) | h32 f32 @196608 (2MB) |
// hT16 f16 @2293760 (1MB) | ha1 @3342336 (16KB) | ha2 @3358720 (16KB) |
// ea2 @3375104 (64B). Total ~3.22 MB. Every byte read is written each call.
// ---------------------------------------------------------------------------
extern "C" void kernel_launch(void* const* d_in, const int* in_sizes, int n_in,
                              void* d_out, int out_size, void* d_ws, size_t ws_size,
                              hipStream_t stream) {
  const float* node_rep = (const float*)d_in[0];
  const float* mask     = (const float*)d_in[1];
  const int*   adj      = (const int*)d_in[2];
  const float* W        = (const float*)d_in[3];
  const float* bias     = (const float*)d_in[4];
  const float* a1       = (const float*)d_in[5];
  const float* a2       = (const float*)d_in[6];
  const float* a_b      = (const float*)d_in[7];
  const float* emb      = (const float*)d_in[8];
  float* out = (float*)d_out;

  char* ws = (char*)d_ws;
  _Float16* Wt   = (_Float16*)(ws + 0);
  float*    h32  = (float*)(ws + 196608);
  _Float16* hT16 = (_Float16*)(ws + 2293760);
  float*    ha1  = (float*)(ws + 3342336);
  float*    ha2  = (float*)(ws + 3358720);
  float*    ea2  = (float*)(ws + 3375104);

  gat_prep  <<<384, 256, 0, stream>>>(W, emb, a2, Wt, ea2);
  gat_proj  <<<256, 256, 0, stream>>>(node_rep, mask, Wt, bias, h32, hT16);
  gat_rowdot<<<512, 256, 0, stream>>>(h32, a1, a2, ha1, ha2);
  gat_attn  <<<256, 128, 0, stream>>>(adj, ha1, ha2, ea2, a_b, hT16, emb, out);
}
